// GCN_62955630624873
// MI455X (gfx1250) — matrix operand order repaired and checked by reading, not compiled, not run on hardware
//
#include <hip/hip_runtime.h>
#include <hip/hip_bf16.h>

// ---------------- CDNA5 WMMA types ----------------
typedef __attribute__((ext_vector_type(16))) __bf16      v16bf;
typedef __attribute__((ext_vector_type(8)))  float       v8f;
typedef __attribute__((ext_vector_type(8)))  unsigned int v8u;

__device__ __forceinline__ unsigned short f32_to_bf16_rne(float f) {
    unsigned int u = __builtin_bit_cast(unsigned int, f);
    unsigned int r = u + 0x7FFFu + ((u >> 16) & 1u);
    return (unsigned short)(r >> 16);
}
__device__ __forceinline__ float bf16_bits_to_f32(unsigned short b) {
    return __builtin_bit_cast(float, (unsigned int)b << 16);
}

// ---------------- degree / normalization ----------------
__global__ void deg_init_k(float* __restrict__ deg, int n) {
    int i = blockIdx.x * blockDim.x + threadIdx.x;
    if (i < n) deg[i] = 1.0f;                       // self-loop
}
__global__ void deg_acc_k(const int* __restrict__ dst, float* __restrict__ deg, int ne) {
    int i = blockIdx.x * blockDim.x + threadIdx.x;
    if (i < ne) atomicAdd(&deg[dst[i]], 1.0f);
}
__global__ void dinv_k(const float* __restrict__ deg, float* __restrict__ dinv, int n) {
    int i = blockIdx.x * blockDim.x + threadIdx.x;
    if (i < n) dinv[i] = rsqrtf(deg[i]);            // deg >= 1 always (self loop)
}

// ---------------- pack W into WMMA B-fragment layout (hi/lo bf16 split) ----
// B (32x16, bf16): lane = column N (0..15 / 16..31 mirrored), VGPR j holds
// K = (j/4)*16 + half*8 + (j&3)*2, +1.  Stored as 16 contiguous dwords per lane:
// [0..7]=hi frag, [8..15]=lo frag.
template<int KDIM>
__global__ void pack_w_k(const float* __restrict__ W, unsigned int* __restrict__ wpk) {
    int kt   = blockIdx.x >> 2;       // K chunk of 32
    int nt   = blockIdx.x & 3;        // N tile of 16
    int lane = threadIdx.x & 31;
    int half = lane >> 4;
    int n    = nt * 16 + (lane & 15);
    unsigned int* o = wpk + (((size_t)kt * 4 + nt) * 32 + lane) * 16;
#pragma unroll
    for (int i = 0; i < 8; ++i) {
        int k = kt * 32 + (i >> 2) * 16 + half * 8 + (i & 3) * 2;
        float f0 = W[(size_t)k * 64 + n];
        float f1 = W[(size_t)(k + 1) * 64 + n];
        unsigned short h0 = f32_to_bf16_rne(f0);
        unsigned short h1 = f32_to_bf16_rne(f1);
        unsigned short l0 = f32_to_bf16_rne(f0 - bf16_bits_to_f32(h0));
        unsigned short l1 = f32_to_bf16_rne(f1 - bf16_bits_to_f32(h1));
        o[i]     = (unsigned int)h0 | ((unsigned int)h1 << 16);
        o[8 + i] = (unsigned int)l0 | ((unsigned int)l1 << 16);
    }
}

// ---------------- GEMM: out[N,64] = act[N,KDIM] @ W, bf16x2 split WMMA ------
// Block = 256 threads = 8 waves; wave w owns rows rowBase..rowBase+15 and all
// 64 output columns (4 N-tiles).  A fragment built per ISA layout:
// lane m / m+16 are row m, K = (i/4)*16 + half*8 + (i&3)*2 (+k0); the two
// lane-halves of a row together consume exactly one 128B line per K-step.
template<int KDIM>
__global__ void __launch_bounds__(256)
gemm_bf16x2_k(const float* __restrict__ act, const unsigned int* __restrict__ wpk,
              float* __restrict__ out, int nrows) {
    const int wave = threadIdx.x >> 5;
    const int lane = threadIdx.x & 31;
    const int half = lane >> 4;
    const int lm   = lane & 15;
    const int rowBase = blockIdx.x * 128 + wave * 16;
    int row  = rowBase + lm;
    int rowc = row < nrows ? row : nrows - 1;       // clamp loads (WMMA needs full EXEC)

    v8f acc[4];
#pragma unroll
    for (int nt = 0; nt < 4; ++nt) acc[nt] = 0;

    const int KIT = KDIM / 32;
    for (int kt = 0; kt < KIT; ++kt) {
        const int k0 = kt * 32;
        v8u ahi, alo;
#pragma unroll
        for (int i = 0; i < 8; ++i) {
            int k = k0 + (i >> 2) * 16 + half * 8 + (i & 3) * 2;
            const float* p = act + (size_t)rowc * KDIM + k;
            float f0 = p[0], f1 = p[1];             // global_load_b64, full line reuse
            unsigned short h0 = f32_to_bf16_rne(f0);
            unsigned short h1 = f32_to_bf16_rne(f1);
            unsigned short l0 = f32_to_bf16_rne(f0 - bf16_bits_to_f32(h0));
            unsigned short l1 = f32_to_bf16_rne(f1 - bf16_bits_to_f32(h1));
            ahi[i] = (unsigned int)h0 | ((unsigned int)h1 << 16);
            alo[i] = (unsigned int)l0 | ((unsigned int)l1 << 16);
        }
        v16bf Ahi = __builtin_bit_cast(v16bf, ahi);
        v16bf Alo = __builtin_bit_cast(v16bf, alo);
#pragma unroll
        for (int nt = 0; nt < 4; ++nt) {
            const unsigned int* bp = wpk + (((size_t)kt * 4 + nt) * 32 + lane) * 16;
            v8u bh, bl;
#pragma unroll
            for (int i = 0; i < 8; ++i) { bh[i] = bp[i]; bl[i] = bp[8 + i]; }
            v16bf Bhi = __builtin_bit_cast(v16bf, bh);
            v16bf Blo = __builtin_bit_cast(v16bf, bl);
            // fp32-accurate product via 3-term bf16 split
            acc[nt] = __builtin_amdgcn_wmma_f32_16x16x32_bf16(
                false, Ahi, false, Bhi, (short)0, acc[nt], false, false);
            acc[nt] = __builtin_amdgcn_wmma_f32_16x16x32_bf16(
                false, Ahi, false, Blo, (short)0, acc[nt], false, false);
            acc[nt] = __builtin_amdgcn_wmma_f32_16x16x32_bf16(
                false, Alo, false, Bhi, (short)0, acc[nt], false, false);
        }
    }
    // C/D layout: VGPR r -> M = r + 8*half, N = lane&15 (+16*nt)
#pragma unroll
    for (int r = 0; r < 8; ++r) {
        int m = rowBase + r + half * 8;
        if (m < nrows) {
            float* o = out + (size_t)m * 64 + lm;
            o[0]  = acc[0][r];
            o[16] = acc[1][r];
            o[32] = acc[2][r];
            o[48] = acc[3][r];
        }
    }
}

// ---------------- self-loop term + bias seeds the accumulator --------------
__global__ void init_self_k(const float* __restrict__ h, const float* __restrict__ dinv,
                            const float* __restrict__ b, float* __restrict__ out, int total) {
    int i = blockIdx.x * blockDim.x + threadIdx.x;
    if (i >= total) return;
    int node = i >> 6, f = i & 63;
    float di = dinv[node];
    out[i] = h[i] * di * di + b[f];
}

// ---------------- edge aggregation: out[dst] += h[src]*norm ----------------
// 16 lanes per edge, float4 gather (256B row = one full line), f32 atomics.
// h and out both fit in the 192MB L2, so the random traffic is L2-resident.
__global__ void agg_edges_k(const float* __restrict__ h, const int* __restrict__ src,
                            const int* __restrict__ dst, const float* __restrict__ dinv,
                            float* __restrict__ out, int ne) {
    long long idx = (long long)blockIdx.x * blockDim.x + threadIdx.x;
    long long e  = idx >> 4;
    int       fq = (int)(idx & 15);
    if (e >= ne) return;
    int s = src[e], d = dst[e];
    float norm = dinv[s] * dinv[d];
    float4 v = ((const float4*)(h + (size_t)s * 64))[fq];
    float* o = out + (size_t)d * 64 + fq * 4;
    atomicAdd(o + 0, v.x * norm);
    atomicAdd(o + 1, v.y * norm);
    atomicAdd(o + 2, v.z * norm);
    atomicAdd(o + 3, v.w * norm);
}

__global__ void relu_k(float* __restrict__ p, int n) {
    int i = blockIdx.x * blockDim.x + threadIdx.x;
    if (i < n) p[i] = fmaxf(p[i], 0.0f);
}

// ---------------- launch -----------------
extern "C" void kernel_launch(void* const* d_in, const int* in_sizes, int n_in,
                              void* d_out, int out_size, void* d_ws, size_t ws_size,
                              hipStream_t stream) {
    const float* x  = (const float*)d_in[0];
    const int*   ei = (const int*)d_in[1];     // int32 (JAX x64 disabled)
    const float* W1 = (const float*)d_in[2];
    const float* b1 = (const float*)d_in[3];
    const float* W2 = (const float*)d_in[4];
    const float* b2 = (const float*)d_in[5];

    const int N = in_sizes[0] / 256;
    const int E = in_sizes[1] / 2;
    const int* src = ei;
    const int* dst = ei + E;

    auto align_up = [](size_t v) { return (v + 255) & ~(size_t)255; };
    char* ws = (char*)d_ws;
    float* deg  = (float*)ws;  ws += align_up((size_t)N * 4);
    float* dinv = (float*)ws;  ws += align_up((size_t)N * 4);
    float* hbuf = (float*)ws;  ws += align_up((size_t)N * 64 * 4);
    float* tbuf = (float*)ws;  ws += align_up((size_t)N * 64 * 4);
    unsigned int* wpk1 = (unsigned int*)ws; ws += align_up((size_t)8 * 4 * 32 * 16 * 4);
    unsigned int* wpk2 = (unsigned int*)ws; ws += align_up((size_t)2 * 4 * 32 * 16 * 4);

    const int T = 256;
    // symmetric normalization
    deg_init_k<<<(N + T - 1) / T, T, 0, stream>>>(deg, N);
    deg_acc_k<<<(E + T - 1) / T, T, 0, stream>>>(dst, deg, E);
    dinv_k<<<(N + T - 1) / T, T, 0, stream>>>(deg, dinv, N);

    // ---- layer 1: x[100000,256] @ W1 -> h1; aggregate; relu ----
    pack_w_k<256><<<32, 32, 0, stream>>>(W1, wpk1);
    gemm_bf16x2_k<256><<<(N + 127) / 128, 256, 0, stream>>>(x, wpk1, hbuf, N);
    init_self_k<<<(N * 64 + T - 1) / T, T, 0, stream>>>(hbuf, dinv, b1, tbuf, N * 64);
    {
        long long tot = (long long)E * 16;
        agg_edges_k<<<(int)((tot + T - 1) / T), T, 0, stream>>>(hbuf, src, dst, dinv, tbuf, E);
    }
    relu_k<<<(N * 64 + T - 1) / T, T, 0, stream>>>(tbuf, N * 64);

    // ---- layer 2: h1[100000,64] @ W2 -> h2; aggregate into d_out; relu ----
    pack_w_k<64><<<8, 32, 0, stream>>>(W2, wpk2);
    gemm_bf16x2_k<64><<<(N + 127) / 128, 256, 0, stream>>>(tbuf, wpk2, hbuf, N);
    float* out = (float*)d_out;
    init_self_k<<<(N * 64 + T - 1) / T, T, 0, stream>>>(hbuf, dinv, b2, out, N * 64);
    {
        long long tot = (long long)E * 16;
        agg_edges_k<<<(int)((tot + T - 1) / T), T, 0, stream>>>(hbuf, src, dst, dinv, out, E);
    }
    relu_k<<<(N * 64 + T - 1) / T, T, 0, stream>>>(out, N * 64);
}
